// SAGENet_2336462209632
// MI455X (gfx1250) — matrix operand order repaired and checked by reading, not compiled, not run on hardware
//
#include <hip/hip_runtime.h>

typedef __attribute__((ext_vector_type(16))) __bf16 v16bf;
typedef __attribute__((ext_vector_type(8)))  __bf16 v8bf;
typedef __attribute__((ext_vector_type(8)))  float  v8f;

#define N_NODES 50000
#define N_EDGES 800000
#define TILES_N 3125            // 50000 / 16, exact

// ---------------------------------------------------------------------------
// Edge scatter: 32 lanes per edge, each lane adds F/32 consecutive features
// with global_atomic_add_f32. Gathers hit L2 (x/h fit in 192MB L2).
// ---------------------------------------------------------------------------
template <int F, bool ADD_CNT>
__global__ void sage_scatter(const float* __restrict__ x,
                             const long long* __restrict__ src,
                             const long long* __restrict__ dst,
                             float* __restrict__ agg,
                             float* __restrict__ cnt) {
    long long tid = (long long)blockIdx.x * blockDim.x + threadIdx.x;
    int edge = (int)(tid >> 5);
    int lane = (int)(tid & 31);
    if (edge >= N_EDGES) return;
    int s = (int)src[edge];
    int d = (int)dst[edge];
    constexpr int PER = F / 32;
    const float* xs = x + (size_t)s * F + lane * PER;
    float* ad = agg + (size_t)d * F + lane * PER;
#pragma unroll
    for (int i = 0; i < PER; ++i) atomicAdd(ad + i, xs[i]);
    if (ADD_CNT && lane == 0) atomicAdd(cnt + d, 1.0f);
}

// ---------------------------------------------------------------------------
// mean + concat + bf16 convert:  Abig[n, 0:F) = agg[n]/max(cnt,1) (bf16)
//                                Abig[n, F:2F) = xin[n]           (bf16)
// ---------------------------------------------------------------------------
template <int F>
__global__ void mean_concat_bf16(const float* __restrict__ agg,
                                 const float* __restrict__ cnt,
                                 const float* __restrict__ xin,
                                 __bf16* __restrict__ Abig) {
    long long tid = (long long)blockIdx.x * blockDim.x + threadIdx.x;
    if (tid >= (long long)N_NODES * F) return;
    int node = (int)(tid / F);
    int f = (int)(tid % F);
    float inv = 1.0f / fmaxf(cnt[node], 1.0f);
    size_t rb = (size_t)node * (2 * F);
    Abig[rb + f]     = (__bf16)(agg[(size_t)node * F + f] * inv);
    Abig[rb + F + f] = (__bf16)(xin[(size_t)node * F + f]);
}

// ---------------------------------------------------------------------------
// Weight prep: stack [Wl; Wr] (each Kh x M row-major fp32) and transpose to
// column-major bf16:  Wt[m * 2Kh + k] = (k<Kh ? Wl[k][m] : Wr[k-Kh][m])
// ---------------------------------------------------------------------------
template <int KH, int M>
__global__ void prep_weights(const float* __restrict__ Wl,
                             const float* __restrict__ Wr,
                             __bf16* __restrict__ Wt) {
    int tid = blockIdx.x * blockDim.x + threadIdx.x;
    if (tid >= M * 2 * KH) return;
    int m = tid / (2 * KH);
    int k = tid % (2 * KH);
    float v = (k < KH) ? Wl[(size_t)k * M + m] : Wr[(size_t)(k - KH) * M + m];
    Wt[tid] = (__bf16)v;
}

// deterministic keep-mask for dropout p=0.1 (seeded integer hash)
__device__ __forceinline__ bool drop_keep(unsigned idx) {
    unsigned h = idx * 2654435761u + 42u;
    h ^= h >> 16; h *= 0x85ebca6bu;
    h ^= h >> 13; h *= 0xc2b2ae35u;
    h ^= h >> 16;
    return (h & 0xFFFFFFu) < 15099494u;   // 0.9 * 2^24
}

// ---------------------------------------------------------------------------
// WMMA GEMM:  out[N x M] = A[N x K](bf16) @ Wt^T + bias, one 16x16 tile/wave.
// A fragment (16x32 bf16): lane<16 -> row lane,  K chunks [k+0..7],[k+8..15]
//                          lane>=16 -> row lane-16, chunks [k+16..23],[k+24..31]
// B fragment: lane owns output column (lane&15), 16 consecutive K values at
//             k + (lane>>4)*16 from column-major Wt.
// C/D: acc[r] -> row ti*16 + r + 8*(lane>>4), col tj*16 + (lane&15).
// ---------------------------------------------------------------------------
template <int K, int M, bool RELU_DROP>
__global__ void gemm_wmma_bf16(const __bf16* __restrict__ A,
                               const __bf16* __restrict__ Wt,
                               const float* __restrict__ bias,
                               float* __restrict__ out) {
    constexpr int TILES_M = M / 16;
    int wave = (int)((blockIdx.x * blockDim.x + threadIdx.x) >> 5);
    int lane = threadIdx.x & 31;
    int ti = wave / TILES_M;
    int tj = wave % TILES_M;
    if (ti >= TILES_N) return;          // wave-uniform: EXEC stays all-ones

    int halfsel = lane >> 4;            // 0: lanes 0-15, 1: lanes 16-31
    const __bf16* Arow = A  + (size_t)(ti * 16 + (lane & 15)) * K + halfsel * 16;
    const __bf16* Wcol = Wt + (size_t)(tj * 16 + (lane & 15)) * K + halfsel * 16;

    v8f acc = {};
#pragma unroll
    for (int k = 0; k < K; k += 32) {
        v8bf a0 = *(const v8bf*)(Arow + k);
        v8bf a1 = *(const v8bf*)(Arow + k + 8);
        v16bf a, b;
        b = *(const v16bf*)(Wcol + k);
#pragma unroll
        for (int i = 0; i < 8; ++i) { a[i] = a0[i]; a[i + 8] = a1[i]; }
        acc = __builtin_amdgcn_wmma_f32_16x16x32_bf16(
            false, a, false, b, (short)0, acc, false, false);
    }

    int col = tj * 16 + (lane & 15);
    int rbase = ti * 16 + (halfsel << 3);
    float bcol = bias[col];
#pragma unroll
    for (int r = 0; r < 8; ++r) {
        int row = rbase + r;
        float v = acc[r] + bcol;
        if (RELU_DROP) {
            v = fmaxf(v, 0.0f);
            v = drop_keep((unsigned)(row * M + col)) ? v * (1.0f / 0.9f) : 0.0f;
        }
        out[(size_t)row * M + col] = v;
    }
}

// ---------------------------------------------------------------------------
// Launch: layer1 scatter -> mean|x concat -> fused WMMA GEMM (+relu+dropout)
//         layer2 scatter -> mean|h concat -> fused WMMA GEMM -> d_out
// ---------------------------------------------------------------------------
extern "C" void kernel_launch(void* const* d_in, const int* in_sizes, int n_in,
                              void* d_out, int out_size, void* d_ws, size_t ws_size,
                              hipStream_t stream) {
    const float*     x   = (const float*)d_in[0];
    const long long* ei  = (const long long*)d_in[1];   // int64 (2, E)
    const float*     Wl1 = (const float*)d_in[2];
    const float*     Wr1 = (const float*)d_in[3];
    const float*     b1  = (const float*)d_in[4];
    const float*     Wl2 = (const float*)d_in[5];
    const float*     Wr2 = (const float*)d_in[6];
    const float*     b2  = (const float*)d_in[7];
    const long long* src = ei;
    const long long* dst = ei + N_EDGES;

    // workspace layout (bytes); agg1+Abig1 region is recycled as agg2
    char* ws = (char*)d_ws;
    const size_t AGG1_B  = (size_t)N_NODES * 128 * 4;   // 25.6 MB
    const size_t AGG2_B  = (size_t)N_NODES * 256 * 4;   // 51.2 MB
    const size_t ABIG2_B = (size_t)N_NODES * 512 * 2;   // 51.2 MB
    const size_t H_B     = (size_t)N_NODES * 256 * 4;   // 51.2 MB

    float*  agg1  = (float*)ws;                          // [0, 25.6M)
    __bf16* Abig1 = (__bf16*)(ws + AGG1_B);              // [25.6M, 51.2M)
    float*  agg2  = (float*)ws;                          // [0, 51.2M)  (reuse)
    __bf16* Abig2 = (__bf16*)(ws + AGG2_B);              // [51.2M, 102.4M)
    float*  h     = (float*)(ws + AGG2_B + ABIG2_B);     // [102.4M, 153.6M)
    float*  cnt   = (float*)(ws + AGG2_B + ABIG2_B + H_B);
    __bf16* Wt1   = (__bf16*)(ws + AGG2_B + ABIG2_B + H_B + 256 * 1024);
    __bf16* Wt2   = Wt1 + 256 * 256;

    // ---- layer 1 -----------------------------------------------------------
    hipMemsetAsync(agg1, 0, AGG1_B, stream);
    hipMemsetAsync(cnt, 0, (size_t)N_NODES * 4, stream);
    prep_weights<128, 256><<<(256 * 256 + 255) / 256, 256, 0, stream>>>(Wl1, Wr1, Wt1);
    prep_weights<256, 128><<<(128 * 512 + 255) / 256, 256, 0, stream>>>(Wl2, Wr2, Wt2);

    {   // scatter: 32 lanes/edge
        long long th = (long long)N_EDGES * 32;
        sage_scatter<128, true><<<(unsigned)((th + 255) / 256), 256, 0, stream>>>(
            x, src, dst, agg1, cnt);
    }
    {   // mean | x -> bf16
        long long th = (long long)N_NODES * 128;
        mean_concat_bf16<128><<<(unsigned)((th + 255) / 256), 256, 0, stream>>>(
            agg1, cnt, x, Abig1);
    }
    {   // h = dropout(relu(Abig1 @ [Wl1;Wr1] + b1)),  waves = 3125*16
        long long th = (long long)TILES_N * 16 * 32;
        gemm_wmma_bf16<256, 256, true><<<(unsigned)((th + 255) / 256), 256, 0, stream>>>(
            Abig1, Wt1, b1, h);
    }

    // ---- layer 2 -----------------------------------------------------------
    hipMemsetAsync(agg2, 0, AGG2_B, stream);
    {
        long long th = (long long)N_EDGES * 32;
        sage_scatter<256, false><<<(unsigned)((th + 255) / 256), 256, 0, stream>>>(
            h, src, dst, agg2, nullptr);
    }
    {
        long long th = (long long)N_NODES * 256;
        mean_concat_bf16<256><<<(unsigned)((th + 255) / 256), 256, 0, stream>>>(
            agg2, cnt, h, Abig2);
    }
    {   // out = Abig2 @ [Wl2;Wr2] + b2,  waves = 3125*8
        long long th = (long long)TILES_N * 8 * 32;
        gemm_wmma_bf16<512, 128, false><<<(unsigned)((th + 255) / 256), 256, 0, stream>>>(
            Abig2, Wt2, b2, (float*)d_out);
    }
    (void)in_sizes; (void)n_in; (void)out_size; (void)ws_size;
}
